// Model_3856880632120
// MI455X (gfx1250) — matrix operand order repaired and checked by reading, not compiled, not run on hardware
//
#include <hip/hip_runtime.h>
#include <hip/hip_bf16.h>
#include <math.h>

// ---------------------------------------------------------------------------
// GPT forward for MI455X (gfx1250): all GEMMs through v_wmma_f32_16x16x32_bf16
// (fp32 accumulate). fp32->bf16 on a register-staged, double-buffered LDS
// pipeline; normal-B tiles are transposed 4x4-in-registers so every LDS store
// is a full-width b64 (no scalar b16 scatter).
// ---------------------------------------------------------------------------

typedef __attribute__((ext_vector_type(4)))  __bf16 bf16x4;
typedef __attribute__((ext_vector_type(8)))  __bf16 bf16x8;
typedef __attribute__((ext_vector_type(16))) __bf16 bf16x16;
typedef __attribute__((ext_vector_type(8)))  float  f32x8;

constexpr int kB  = 2;
constexpr int kT  = 1024;
constexpr int kD  = 1024;
constexpr int kL  = 8;
constexpr int kV  = 50257;
constexpr int kH  = 4096;          // 4*D
constexpr int BT  = kB * kT;       // 2048 rows of activations

constexpr int LDS_STRIDE = 40;     // bf16 elems per LDS row (80 B, 16B aligned)

__device__ __forceinline__ __bf16 to_bf16(float f) { return (__bf16)f; }

// ---------------------------------------------------------------------------
// Generic bf16 WMMA GEMM:  C[M,N] = op( A[M,K] @ B[K,N] (+ bias) )
//   TRANSB : B supplied row-major [N,K] (logical B = Bsrc^T)
//   OP     : 0 = none, 1 = SiLU
//   NGUARD : bounds-check N (only the lm-head needs it; V = 50257)
// Block tile 128x128x32, 256 threads (8 wave32), wave tile 32x64 (2x4 WMMAs).
// M, K assumed multiples of 128 / 32 (true for every call here).
// ---------------------------------------------------------------------------
template<bool TRANSB, int OP, bool HASBIAS, bool NGUARD>
__global__ __launch_bounds__(256) void gemm_bf16_kernel(
    const float* __restrict__ A, const float* __restrict__ Bm,
    const float* __restrict__ bias, float* __restrict__ C,
    int M, int N, int K, int lda, int ldb, int ldc,
    long long sA, long long sB, long long sC)
{
  __shared__ __attribute__((aligned(16))) __bf16 Alds[2][128 * LDS_STRIDE];
  __shared__ __attribute__((aligned(16))) __bf16 Blds[2][128 * LDS_STRIDE];

  const int tid = threadIdx.x;
  const int z   = blockIdx.z;
  A  += (long long)z * sA;
  Bm += (long long)z * sB;
  C  += (long long)z * sC;

  const int n0 = blockIdx.x * 128;
  const int m0 = blockIdx.y * 128;

  const int lane = tid & 31;
  const int wid  = tid >> 5;
  const int wm   = wid & 3;        // 4 waves along M
  const int wn   = wid >> 2;       // 2 waves along N
  const int half = lane >> 4;      // 0: lanes 0-15, 1: lanes 16-31
  const int lrow = lane & 15;

  // staging coordinates
  const int aRow = tid >> 3;             // +p*32 rows (A, TRANSB-B)
  const int aC4  = (tid & 7) * 4;        // K offset within row
  const int bN4  = (tid & 31) * 4;       // normal B: 4 cols   n4..n4+3
  const int bKb  = (tid >> 5) * 4;       // normal B: 4 k-rows kb..kb+3

  f32x8 acc[2][4];
#pragma unroll
  for (int i = 0; i < 2; ++i)
#pragma unroll
    for (int j = 0; j < 4; ++j)
#pragma unroll
      for (int r = 0; r < 8; ++r) acc[i][j][r] = 0.0f;

  // Fragment element offsets inside an LDS buffer (two 16B chunks each):
  //   A 16x32 bf16: lane row M=lrow; chunks at bytes half*16 and 32+half*16
  //   B 32x16 bf16: lane col N=lrow; chunks at bytes half*16 and 32+half*16
  int aOff[2], bOff[4];
#pragma unroll
  for (int i = 0; i < 2; ++i)
    aOff[i] = (wm * 32 + i * 16 + lrow) * LDS_STRIDE + half * 8;
#pragma unroll
  for (int j = 0; j < 4; ++j)
    bOff[j] = (wn * 64 + j * 16 + lrow) * LDS_STRIDE + half * 8;

  float4 ra[4], rb[4];

  // ---- tile loaders (issue loads; no dependent ops -> overlappable) --------
  auto loadA = [&](int k0) {
#pragma unroll
    for (int p = 0; p < 4; ++p)
      ra[p] = *(const float4*)(A + (long long)(m0 + aRow + p * 32) * lda + k0 + aC4);
  };
  auto loadB = [&](int k0) {
    if (TRANSB) {
#pragma unroll
      for (int p = 0; p < 4; ++p) {
        int r = aRow + p * 32;
        if (!NGUARD || (n0 + r < N))
          rb[p] = *(const float4*)(Bm + (long long)(n0 + r) * ldb + k0 + aC4);
        else
          rb[p] = make_float4(0.f, 0.f, 0.f, 0.f);
      }
    } else {
      // 4x4 patch: rows k0+bKb..+3, cols n0+bN4..+3 (coalesced across lanes)
#pragma unroll
      for (int r = 0; r < 4; ++r) {
        if (!NGUARD || (n0 + bN4 < N))
          rb[r] = *(const float4*)(Bm + (long long)(k0 + bKb + r) * ldb + n0 + bN4);
        else
          rb[r] = make_float4(0.f, 0.f, 0.f, 0.f);
      }
    }
  };
  auto storeTiles = [&](int buf) {
#pragma unroll
    for (int p = 0; p < 4; ++p) {
      __bf16* dst = &Alds[buf][(aRow + p * 32) * LDS_STRIDE + aC4];
      dst[0] = to_bf16(ra[p].x); dst[1] = to_bf16(ra[p].y);
      dst[2] = to_bf16(ra[p].z); dst[3] = to_bf16(ra[p].w);
    }
    if (TRANSB) {
#pragma unroll
      for (int p = 0; p < 4; ++p) {
        __bf16* dst = &Blds[buf][(aRow + p * 32) * LDS_STRIDE + aC4];
        dst[0] = to_bf16(rb[p].x); dst[1] = to_bf16(rb[p].y);
        dst[2] = to_bf16(rb[p].z); dst[3] = to_bf16(rb[p].w);
      }
    } else {
      // register 4x4 transpose -> 4 contiguous-K b64 stores into Blds[n][k]
#pragma unroll
      for (int c = 0; c < 4; ++c) {
        bf16x4 t;
        t[0] = to_bf16(((const float*)&rb[0])[c]);
        t[1] = to_bf16(((const float*)&rb[1])[c]);
        t[2] = to_bf16(((const float*)&rb[2])[c]);
        t[3] = to_bf16(((const float*)&rb[3])[c]);
        *(bf16x4*)(&Blds[buf][(bN4 + c) * LDS_STRIDE + bKb]) = t;
      }
    }
  };

  // ---- software-pipelined main loop (one barrier per K-step) ---------------
  loadA(0);
  loadB(0);
  int buf = 0;
  for (int k0 = 0; k0 < K; k0 += 32) {
    storeTiles(buf);
    __syncthreads();
    if (k0 + 32 < K) {        // next tile's loads fly under the WMMAs
      loadA(k0 + 32);
      loadB(k0 + 32);
    }
    bf16x16 af[2], bfr[4];
#pragma unroll
    for (int i = 0; i < 2; ++i) {
      bf16x8 lo = *(const bf16x8*)(&Alds[buf][aOff[i]]);
      bf16x8 hi = *(const bf16x8*)(&Alds[buf][aOff[i] + 16]);
#pragma unroll
      for (int e = 0; e < 8; ++e) { af[i][e] = lo[e]; af[i][e + 8] = hi[e]; }
    }
#pragma unroll
    for (int j = 0; j < 4; ++j) {
      bf16x8 lo = *(const bf16x8*)(&Blds[buf][bOff[j]]);
      bf16x8 hi = *(const bf16x8*)(&Blds[buf][bOff[j] + 16]);
#pragma unroll
      for (int e = 0; e < 8; ++e) { bfr[j][e] = lo[e]; bfr[j][e + 8] = hi[e]; }
    }
#pragma unroll
    for (int i = 0; i < 2; ++i)
#pragma unroll
      for (int j = 0; j < 4; ++j)
        acc[i][j] = __builtin_amdgcn_wmma_f32_16x16x32_bf16(
            false, af[i], false, bfr[j], (short)0, acc[i][j], false, false);
    buf ^= 1;
  }

  // ---- epilogue: C/D layout lane lrow = N, VGPR r -> M = r + 8*half --------
#pragma unroll
  for (int i = 0; i < 2; ++i) {
#pragma unroll
    for (int j = 0; j < 4; ++j) {
      int n_g = n0 + wn * 64 + j * 16 + lrow;
      if (!NGUARD || (n_g < N)) {
        float bv = HASBIAS ? bias[n_g] : 0.0f;
#pragma unroll
        for (int r = 0; r < 8; ++r) {
          int m_g = m0 + wm * 32 + i * 16 + half * 8 + r;
          float v = acc[i][j][r] + bv;
          if (OP == 1) v = v / (1.0f + __expf(-v));   // SiLU
          C[(long long)m_g * ldc + n_g] = v;
        }
      }
    }
  }
}

// ---------------------------------------------------------------------------
// LayerNorm (per row of D=1024): y = (x-mean)*rsqrt(var+eps)*w (+ b)
// ---------------------------------------------------------------------------
__global__ __launch_bounds__(256) void ln_kernel(
    const float* __restrict__ x, const float* __restrict__ w,
    const float* __restrict__ b, float* __restrict__ y, int D)
{
  __shared__ float red[256];
  const int row = blockIdx.x, tid = threadIdx.x;
  const float* xr = x + (long long)row * D;
  float s = 0.f, ss = 0.f;
  for (int d = tid; d < D; d += 256) { float v = xr[d]; s += v; ss += v * v; }
  red[tid] = s; __syncthreads();
  for (int o = 128; o > 0; o >>= 1) { if (tid < o) red[tid] += red[tid + o]; __syncthreads(); }
  float mean = red[0] / (float)D; __syncthreads();
  red[tid] = ss; __syncthreads();
  for (int o = 128; o > 0; o >>= 1) { if (tid < o) red[tid] += red[tid + o]; __syncthreads(); }
  float var  = red[0] / (float)D - mean * mean;
  float rstd = rsqrtf(var + 1e-5f);
  float* yr = y + (long long)row * D;
  for (int d = tid; d < D; d += 256) {
    float v = (xr[d] - mean) * rstd * w[d];
    if (b) v += b[d];
    yr[d] = v;
  }
}

// ---------------------------------------------------------------------------
// Causal softmax over attention logits, in place; folds the 1/sqrt(D) scale.
// One block per (batch, t) row; s > t set to 0 (exp(-inf)).
// ---------------------------------------------------------------------------
__global__ __launch_bounds__(256) void softmax_causal_kernel(float* __restrict__ wei)
{
  __shared__ float red[256];
  const int row = blockIdx.x, tid = threadIdx.x;
  const int batch = row / kT, t = row % kT;
  float* p = wei + (long long)batch * kT * kT + (long long)t * kT;
  const float scale = 0.03125f;  // 1024^-0.5
  float m = -3.4e38f;
  for (int s_ = tid; s_ <= t; s_ += 256) m = fmaxf(m, p[s_] * scale);
  red[tid] = m; __syncthreads();
  for (int o = 128; o > 0; o >>= 1) { if (tid < o) red[tid] = fmaxf(red[tid], red[tid + o]); __syncthreads(); }
  m = red[0]; __syncthreads();
  float sum = 0.f;
  for (int s_ = tid; s_ <= t; s_ += 256) sum += __expf(p[s_] * scale - m);
  red[tid] = sum; __syncthreads();
  for (int o = 128; o > 0; o >>= 1) { if (tid < o) red[tid] += red[tid + o]; __syncthreads(); }
  sum = red[0];
  const float inv = 1.0f / sum;
  for (int s_ = tid; s_ < kT; s_ += 256)
    p[s_] = (s_ <= t) ? __expf(p[s_] * scale - m) * inv : 0.0f;
}

// ---------------------------------------------------------------------------
__global__ void embed_kernel(const float* __restrict__ tok_emb,
                             const float* __restrict__ pos_emb,
                             const int* __restrict__ x, float* __restrict__ h)
{
  const int idx = blockIdx.x, tid = threadIdx.x;
  const int t   = idx % kT;
  const int tok = x[idx];
  const float* te = tok_emb + (long long)tok * kD;
  const float* pe = pos_emb + (long long)t * kD;
  float* hr = h + (long long)idx * kD;
  for (int d = tid; d < kD; d += 256) hr[d] = te[d] + pe[d];
}

__global__ void add_kernel(float* __restrict__ y, const float* __restrict__ x, int n)
{
  int i = blockIdx.x * 256 + threadIdx.x;
  if (i < n) y[i] += x[i];
}

__global__ void zero_kernel(float* p) { if (threadIdx.x == 0) *p = 0.0f; }

// loss_i = logsumexp(logits_row) - logit[target]; mean over rows into *loss
__global__ __launch_bounds__(256) void loss_kernel(
    const float* __restrict__ logits, const int* __restrict__ targets,
    float* __restrict__ loss)
{
  __shared__ float red[256];
  const int row = blockIdx.x, tid = threadIdx.x;
  const float* p = logits + (long long)row * kV;
  float m = -3.4e38f;
  for (int j = tid; j < kV; j += 256) m = fmaxf(m, p[j]);
  red[tid] = m; __syncthreads();
  for (int o = 128; o > 0; o >>= 1) { if (tid < o) red[tid] = fmaxf(red[tid], red[tid + o]); __syncthreads(); }
  m = red[0]; __syncthreads();
  float sum = 0.f;
  for (int j = tid; j < kV; j += 256) sum += __expf(p[j] - m);
  red[tid] = sum; __syncthreads();
  for (int o = 128; o > 0; o >>= 1) { if (tid < o) red[tid] += red[tid + o]; __syncthreads(); }
  if (tid == 0) {
    float li = logf(red[0]) + m - p[targets[row]];
    atomicAdd(loss, li * (1.0f / (float)BT));
  }
}

// ---------------------------------------------------------------------------
extern "C" void kernel_launch(void* const* d_in, const int* in_sizes, int n_in,
                              void* d_out, int out_size, void* d_ws, size_t ws_size,
                              hipStream_t stream)
{
  const int*   x         = (const int*)  d_in[0];
  const int*   targets   = (const int*)  d_in[1];
  const float* tok_emb   = (const float*)d_in[2];
  const float* pos_emb   = (const float*)d_in[3];
  const float* norm1_w   = (const float*)d_in[4];
  const float* wq        = (const float*)d_in[5];
  const float* wk        = (const float*)d_in[6];
  const float* wv        = (const float*)d_in[7];
  const float* norm2_w   = (const float*)d_in[8];
  const float* fc1       = (const float*)d_in[9];
  const float* c_proj    = (const float*)d_in[10];
  const float* lnf_w     = (const float*)d_in[11];
  const float* lnf_b     = (const float*)d_in[12];
  const float* lm_head_b = (const float*)d_in[13];

  float* out    = (float*)d_out;
  float* logits = out;                              // [BT, V]
  float* loss   = out + (long long)BT * kV;         // [1]

  // workspace layout (floats)
  const long long NTD = (long long)BT * kD;         // 2,097,152
  float* ws  = (float*)d_ws;
  float* h   = ws;                 // [BT, D]
  float* xn  = h   + NTD;          // [BT, D]
  float* qb  = xn  + NTD;          // [BT, D]
  float* kb  = qb  + NTD;          // [BT, D]
  float* vb  = kb  + NTD;          // [BT, D]
  float* wei = vb  + NTD;          // [B, T, T]  (== NTD floats)
  float* t1  = wei + NTD;          // [BT, 4D]
  float* t2  = t1  + (long long)BT * kH;  // [BT, D]

  const dim3 blk(256);
  const long long TD = (long long)kT * kD;
  const long long TT = (long long)kT * kT;

  embed_kernel<<<BT, 256, 0, stream>>>(tok_emb, pos_emb, x, h);

  for (int l = 0; l < kL; ++l) {
    ln_kernel<<<BT, 256, 0, stream>>>(h, norm1_w + (long long)l * kD, nullptr, xn, kD);

    // q, k, v = xn @ W  (W stored [in,out] row-major -> normal B)
    gemm_bf16_kernel<false, 0, false, false><<<dim3(kD / 128, BT / 128, 1), blk, 0, stream>>>(
        xn, wq + (long long)l * kD * kD, nullptr, qb,
        BT, kD, kD, kD, kD, kD, 0, 0, 0);
    gemm_bf16_kernel<false, 0, false, false><<<dim3(kD / 128, BT / 128, 1), blk, 0, stream>>>(
        xn, wk + (long long)l * kD * kD, nullptr, kb,
        BT, kD, kD, kD, kD, kD, 0, 0, 0);
    gemm_bf16_kernel<false, 0, false, false><<<dim3(kD / 128, BT / 128, 1), blk, 0, stream>>>(
        xn, wv + (long long)l * kD * kD, nullptr, vb,
        BT, kD, kD, kD, kD, kD, 0, 0, 0);

    // wei[b] = q[b] @ k[b]^T   (TRANSB, batched over z)
    gemm_bf16_kernel<true, 0, false, false><<<dim3(kT / 128, kT / 128, kB), blk, 0, stream>>>(
        qb, kb, nullptr, wei, kT, kT, kD, kD, kD, kT, TD, TD, TT);

    softmax_causal_kernel<<<BT, 256, 0, stream>>>(wei);

    // attn_out[b] = wei[b] @ v[b]
    gemm_bf16_kernel<false, 0, false, false><<<dim3(kD / 128, kT / 128, kB), blk, 0, stream>>>(
        wei, vb, nullptr, t2, kT, kD, kT, kT, kD, kD, TT, TD, TD);

    add_kernel<<<(int)(NTD / 256), 256, 0, stream>>>(h, t2, (int)NTD);

    ln_kernel<<<BT, 256, 0, stream>>>(h, norm2_w + (long long)l * kD, nullptr, xn, kD);

    // t1 = silu(xn @ fc1[l])
    gemm_bf16_kernel<false, 1, false, false><<<dim3(kH / 128, BT / 128, 1), blk, 0, stream>>>(
        xn, fc1 + (long long)l * kD * kH, nullptr, t1,
        BT, kH, kD, kD, kH, kH, 0, 0, 0);
    // t2 = t1 @ c_proj[l]
    gemm_bf16_kernel<false, 0, false, false><<<dim3(kD / 128, BT / 128, 1), blk, 0, stream>>>(
        t1, c_proj + (long long)l * kH * kD, nullptr, t2,
        BT, kD, kH, kH, kD, kD, 0, 0, 0);

    add_kernel<<<(int)(NTD / 256), 256, 0, stream>>>(h, t2, (int)NTD);
  }

  ln_kernel<<<BT, 256, 0, stream>>>(h, lnf_w, lnf_b, xn, kD);

  // logits = xn @ tok_emb^T + lm_head_b   (tied weights, TRANSB, N guarded)
  gemm_bf16_kernel<true, 0, true, true><<<dim3((kV + 127) / 128, BT / 128, 1), blk, 0, stream>>>(
      xn, tok_emb, lm_head_b, logits, BT, kV, kD, kD, kD, kV, 0, 0, 0);

  zero_kernel<<<1, 1, 0, stream>>>(loss);
  loss_kernel<<<BT, 256, 0, stream>>>(logits, targets, loss);
}
